// Transformer_30081950941633
// MI455X (gfx1250) — matrix operand order repaired and checked by reading, not compiled, not run on hardware
//
#include <hip/hip_runtime.h>
#include <stdint.h>

// ---------------------------------------------------------------------------
// Problem constants (match reference)
// ---------------------------------------------------------------------------
constexpr int kD   = 512;
constexpr int kH   = 8;
constexpr int kDK  = 64;
constexpr int kDFF = 2048;
constexpr int kV   = 8;
constexpr int kU   = 48;
constexpr int kB   = 128;
constexpr int kSU  = 11 + kV;         // 19
constexpr int kTGT = 97;
constexpr int kM1  = kU * kB * kSU;   // 116736 user tokens
constexpr int kM2  = kB * kU;         // 6144 main tokens

typedef __attribute__((ext_vector_type(16))) _Float16 v16h;
typedef __attribute__((ext_vector_type(8)))  float    v8f;
typedef int vi4 __attribute__((vector_size(16)));      // matches builtin param type

// ---------------------------------------------------------------------------
// CDNA5 async global->LDS path (ASYNCcnt-tracked DMA into LDS)
// ---------------------------------------------------------------------------
#if defined(__has_builtin)
#  if __has_builtin(__builtin_amdgcn_global_load_async_to_lds_b128) && \
      __has_builtin(__builtin_amdgcn_s_wait_asynccnt)
#    define USE_ASYNC_LDS 1
#  endif
#endif
#ifndef USE_ASYNC_LDS
#  define USE_ASYNC_LDS 0
#endif

#if USE_ASYNC_LDS
__device__ __forceinline__ void async_b128(const _Float16* g, _Float16* l) {
    // global flat ptr == AS(1) value; AS(3) ptr is 32-bit, and the low 32 bits
    // of a flat LDS address are the LDS offset (ISA 10.2), so uintptr_t casts
    // produce the right values without address-space cast restrictions.
    __builtin_amdgcn_global_load_async_to_lds_b128(
        (__attribute__((address_space(1))) vi4*)(uintptr_t)g,
        (__attribute__((address_space(3))) vi4*)(uintptr_t)l,
        0, 0);
}
#endif

// ---------------------------------------------------------------------------
// fp32 [K,N] -> fp16 [N,K] weight transpose-convert (once per launch, off the
// hot path; lets the GEMM stage B tiles with contiguous b128 LDS stores)
// ---------------------------------------------------------------------------
__global__ void tcvt_kernel(const float* __restrict__ src, _Float16* __restrict__ dst,
                            int K, int N, long total) {
    long id = (long)blockIdx.x * blockDim.x + threadIdx.x;
    if (id >= total) return;
    const long per = (long)K * N;
    const long l = id / per, rem = id % per;
    const int n = (int)(rem / K), k = (int)(rem % K);
    dst[l * per + (long)n * K + k] = (_Float16)src[l * per + (long)k * N + n];
}

// ---------------------------------------------------------------------------
// Embedding gather: build tokens [U*B, 19, D] in f32 master + f16 copy
// ---------------------------------------------------------------------------
__global__ __launch_bounds__(512)
void embed_kernel(const int* __restrict__ sd, const int* __restrict__ ld_,
                  const int* __restrict__ st, const int* __restrict__ sv,
                  const int* __restrict__ rt, const int* __restrict__ pw,
                  const int* __restrict__ dw, const int* __restrict__ vid,
                  const int* __restrict__ fl, const int* __restrict__ dist,
                  const float* __restrict__ e_sd, const float* __restrict__ e_load,
                  const float* __restrict__ e_status, const float* __restrict__ e_served,
                  const float* __restrict__ e_rt, const float* __restrict__ e_time,
                  const float* __restrict__ e_vid, const float* __restrict__ e_flag,
                  const float* __restrict__ e_dist,
                  float* __restrict__ Xf, _Float16* __restrict__ XH) {
    const int rr = blockIdx.x;            // token row = (u*B+b)*19 + t
    const int t  = rr % kSU;
    const int ub = rr / kSU;              // u*B + b
    const int u  = ub / kB, b = ub % kB;
    const float* src;
    if      (t == 0)  src = e_sd     + (size_t)sd[ub] * kD;
    else if (t == 1)  src = e_load   + (size_t)ld_[ub] * kD;
    else if (t == 2)  src = e_status + (size_t)st[ub] * kD;
    else if (t == 3)  src = e_served + (size_t)sv[ub] * kD;
    else if (t == 4)  src = e_rt     + (size_t)rt[ub] * kD;
    else if (t == 5)  src = e_time   + (size_t)pw[ub*2+0] * kD;
    else if (t == 6)  src = e_time   + (size_t)pw[ub*2+1] * kD;
    else if (t == 7)  src = e_time   + (size_t)dw[ub*2+0] * kD;
    else if (t == 8)  src = e_time   + (size_t)dw[ub*2+1] * kD;
    else if (t == 9)  src = e_vid    + (size_t)vid[ub] * kD;
    else if (t == 10) src = e_flag   + (size_t)fl[ub] * kD;
    else { int k = t - 11; src = e_dist + (size_t)dist[(u*kV + k)*kB + b] * kD; }
    const int d = threadIdx.x;
    const float v = src[d];
    Xf[(size_t)rr * kD + d] = v;
    XH[(size_t)rr * kD + d] = (_Float16)v;
}

// ---------------------------------------------------------------------------
// WMMA GEMM: C[M,N] = A[M,K] @ W[K,N] (+bias, opt ReLU), f16 in / f32 acc.
// W is pre-transposed to [N,K]. Block 256 thr (8 waves), tile 128x128, BK=32.
// Wave tile 32x64 = 2x4 WMMA. Double-buffered LDS; async DMA staging when the
// toolchain exposes global_load_async_to_lds, else batched register staging.
// ---------------------------------------------------------------------------
constexpr int BM = 128, BN = 128, BK = 32, KP = 48;  // KP: padded K stride (96B rows)

__device__ __forceinline__ v16h frag_a(const _Float16* s, int row0, int ld) {
    const int lane = threadIdx.x & 31;
    const int m = lane & 15, h = lane >> 4;
    const _Float16* p = s + (row0 + m) * ld;
    union { v16h v; uint4 u[2]; } f;
    f.u[0] = *reinterpret_cast<const uint4*>(p + 16 * h);      // K = 16h..16h+7
    f.u[1] = *reinterpret_cast<const uint4*>(p + 16 * h + 8);  // K = 16h+8..16h+15
    return f.v;
}
__device__ __forceinline__ v16h frag_b(const _Float16* s, int col0, int ld) {
    const int lane = threadIdx.x & 31;
    const int n = lane & 15, h = lane >> 4;
    const _Float16* p = s + (col0 + n) * ld + 16 * h;          // K = 16h..16h+15
    union { v16h v; uint4 u[2]; } f;
    f.u[0] = *reinterpret_cast<const uint4*>(p);
    f.u[1] = *reinterpret_cast<const uint4*>(p + 8);
    return f.v;
}

template <bool RELU, bool OUTF32, bool OUTF16>
__global__ __launch_bounds__(256)
void gemm_wmma(const _Float16* __restrict__ A, const _Float16* __restrict__ Wt,
               const float* __restrict__ bias, float* __restrict__ Cf,
               _Float16* __restrict__ Ch, int M, int N, int K) {
    __shared__ _Float16 As[2][BM * KP];
    __shared__ _Float16 Bs[2][BN * KP];
    const int tid = threadIdx.x, lane = tid & 31, wave = tid >> 5;
    const int bm = blockIdx.x * BM, bn = blockIdx.y * BN;
    const int wm = (wave & 3) * 32, wn = (wave >> 2) * 64;

    // Staging: thread -> rows (sr, sr+64), 8-half chunk sk. 4 b128 moves/tile.
    const int sr = tid >> 2;               // 0..63
    const int sk = (tid & 3) << 3;         // 0,8,16,24
    const _Float16* ga = A  + (size_t)(bm + sr) * K + sk;
    const _Float16* gb = Wt + (size_t)(bn + sr) * K + sk;
    const size_t rskip = (size_t)64 * K;
    const int l0 = sr * KP + sk, l1 = (sr + 64) * KP + sk;

    v8f acc[2][4];
#pragma unroll
    for (int i = 0; i < 2; ++i)
#pragma unroll
        for (int j = 0; j < 4; ++j)
#pragma unroll
            for (int r = 0; r < 8; ++r) acc[i][j][r] = 0.0f;

    int buf = 0;
#if USE_ASYNC_LDS
    async_b128(ga,         &As[0][l0]);
    async_b128(ga + rskip, &As[0][l1]);
    async_b128(gb,         &Bs[0][l0]);
    async_b128(gb + rskip, &Bs[0][l1]);
#else
    uint4 ra0 = *reinterpret_cast<const uint4*>(ga);
    uint4 ra1 = *reinterpret_cast<const uint4*>(ga + rskip);
    uint4 rb0 = *reinterpret_cast<const uint4*>(gb);
    uint4 rb1 = *reinterpret_cast<const uint4*>(gb + rskip);
    *reinterpret_cast<uint4*>(&As[0][l0]) = ra0;
    *reinterpret_cast<uint4*>(&As[0][l1]) = ra1;
    *reinterpret_cast<uint4*>(&Bs[0][l0]) = rb0;
    *reinterpret_cast<uint4*>(&Bs[0][l1]) = rb1;
#endif

    for (int k0 = 0; k0 < K; k0 += BK) {
        const bool next = (k0 + BK) < K;
#if USE_ASYNC_LDS
        __builtin_amdgcn_s_wait_asynccnt(0);   // this wave's DMA into LDS done
        __syncthreads();                       // everyone's tile visible
        if (next) {                            // DMA tile k0+BK into other buffer
            async_b128(ga + (k0 + BK),         &As[buf ^ 1][l0]);
            async_b128(ga + rskip + (k0 + BK), &As[buf ^ 1][l1]);
            async_b128(gb + (k0 + BK),         &Bs[buf ^ 1][l0]);
            async_b128(gb + rskip + (k0 + BK), &Bs[buf ^ 1][l1]);
        }
#else
        __syncthreads();
        if (next) {                            // issue loads early, store after compute
            ra0 = *reinterpret_cast<const uint4*>(ga + (k0 + BK));
            ra1 = *reinterpret_cast<const uint4*>(ga + rskip + (k0 + BK));
            rb0 = *reinterpret_cast<const uint4*>(gb + (k0 + BK));
            rb1 = *reinterpret_cast<const uint4*>(gb + rskip + (k0 + BK));
        }
#endif
        if (k0 + 2 * BK < K) {                 // keep far prefetch in flight
            __builtin_prefetch(ga + (k0 + 2 * BK), 0, 1);
            __builtin_prefetch(gb + (k0 + 2 * BK), 0, 1);
        }

        const _Float16* as = &As[buf][0];
        const _Float16* bs = &Bs[buf][0];
        v16h af0 = frag_a(as, wm, KP);
        v16h af1 = frag_a(as, wm + 16, KP);
        v16h bf[4];
#pragma unroll
        for (int j = 0; j < 4; ++j) bf[j] = frag_b(bs, wn + 16 * j, KP);
#pragma unroll
        for (int j = 0; j < 4; ++j) {
            acc[0][j] = __builtin_amdgcn_wmma_f32_16x16x32_f16(
                false, af0, false, bf[j], (short)0, acc[0][j], false, false);
            acc[1][j] = __builtin_amdgcn_wmma_f32_16x16x32_f16(
                false, af1, false, bf[j], (short)0, acc[1][j], false, false);
        }

#if !USE_ASYNC_LDS
        if (next) {
            *reinterpret_cast<uint4*>(&As[buf ^ 1][l0]) = ra0;
            *reinterpret_cast<uint4*>(&As[buf ^ 1][l1]) = ra1;
            *reinterpret_cast<uint4*>(&Bs[buf ^ 1][l0]) = rb0;
            *reinterpret_cast<uint4*>(&Bs[buf ^ 1][l1]) = rb1;
        }
#endif
        buf ^= 1;
    }

    // Epilogue: VGPR r of a 16x16 f32 tile holds M = r (+8 on upper half-wave)
    const int mh = (lane >> 4) * 8;
    const int nn = lane & 15;
#pragma unroll
    for (int i = 0; i < 2; ++i)
#pragma unroll
        for (int j = 0; j < 4; ++j) {
            const int col = bn + wn + 16 * j + nn;
            const float bv = bias ? bias[col] : 0.0f;
#pragma unroll
            for (int r = 0; r < 8; ++r) {
                const int row = bm + wm + 16 * i + mh + r;
                float v = acc[i][j][r] + bv;
                if (RELU) v = fmaxf(v, 0.0f);
                const size_t idx = (size_t)row * N + col;
                if (OUTF32) Cf[idx] = v;
                if (OUTF16) Ch[idx] = (_Float16)v;
            }
        }
}

// ---------------------------------------------------------------------------
// Attention: one block per (sequence, head). <1% of FLOPs -> VALU is fine.
// ---------------------------------------------------------------------------
__global__ __launch_bounds__(128)
void attn_kernel(const _Float16* __restrict__ Q, const _Float16* __restrict__ K,
                 const _Float16* __restrict__ V, _Float16* __restrict__ AO, int S) {
    extern __shared__ float smem[];
    float* qs = smem;
    float* ks = qs + S * kDK;
    float* vs = ks + S * kDK;
    float* sc = vs + S * kDK;
    const int seq = blockIdx.x >> 3;   // / H
    const int head = blockIdx.x & 7;
    const int tid = threadIdx.x;
    const size_t rowbase = (size_t)seq * S;

    for (int p = tid; p < S * kDK; p += 128) {
        const int s = p >> 6, d = p & 63;
        const size_t g = (rowbase + s) * kD + head * kDK + d;
        qs[p] = (float)Q[g];
        ks[p] = (float)K[g];
        vs[p] = (float)V[g];
    }
    __syncthreads();

    for (int p = tid; p < S * S; p += 128) {
        const int i = p / S, j = p % S;
        float a = 0.0f;
#pragma unroll 16
        for (int d = 0; d < kDK; ++d) a += qs[i * kDK + d] * ks[j * kDK + d];
        sc[p] = a * 0.125f;            // / sqrt(DK)
    }
    __syncthreads();

    const int wave = tid >> 5, lane = tid & 31;   // wave32
    for (int i = wave; i < S; i += 4) {
        float mx = -3.0e38f;
        for (int j = lane; j < S; j += 32) mx = fmaxf(mx, sc[i * S + j]);
#pragma unroll
        for (int o = 16; o; o >>= 1) mx = fmaxf(mx, __shfl_xor(mx, o));
        float sum = 0.0f;
        for (int j = lane; j < S; j += 32) {
            const float e = __expf(sc[i * S + j] - mx);
            sc[i * S + j] = e;
            sum += e;
        }
#pragma unroll
        for (int o = 16; o; o >>= 1) sum += __shfl_xor(sum, o);
        const float inv = 1.0f / sum;
        for (int j = lane; j < S; j += 32) sc[i * S + j] *= inv;
    }
    __syncthreads();

    for (int p = tid; p < S * kDK; p += 128) {
        const int i = p >> 6, d = p & 63;
        float a = 0.0f;
        for (int j = 0; j < S; ++j) a += sc[i * S + j] * vs[j * kDK + d];
        AO[(rowbase + i) * kD + head * kDK + d] = (_Float16)a;
    }
}

// ---------------------------------------------------------------------------
// Residual + LayerNorm over the LAST TWO dims (S*D per sequence), affine.
// One block per sequence. Updates f32 master and f16 GEMM copy.
// ---------------------------------------------------------------------------
__global__ __launch_bounds__(256)
void ln_kernel(float* __restrict__ Xf, _Float16* __restrict__ XH,
               const float* __restrict__ Hres, const float* __restrict__ g,
               const float* __restrict__ be, int S) {
    __shared__ float r1[256], r2[256];
    const int tid = threadIdx.x;
    const int n = S * kD;
    const size_t base = (size_t)blockIdx.x * n;
    float s = 0.0f, ss = 0.0f;
    for (int e = tid; e < n; e += 256) {
        const float v = Xf[base + e] + Hres[base + e];
        s += v; ss += v * v;
    }
    r1[tid] = s; r2[tid] = ss;
    __syncthreads();
    for (int o = 128; o > 0; o >>= 1) {
        if (tid < o) { r1[tid] += r1[tid + o]; r2[tid] += r2[tid + o]; }
        __syncthreads();
    }
    const float mean = r1[0] / n;
    const float var  = r2[0] / n - mean * mean;
    const float inv  = rsqrtf(var + 1e-6f);
    for (int e = tid; e < n; e += 256) {
        const float v = (Xf[base + e] + Hres[base + e] - mean) * inv;
        const float o = v * g[e] + be[e];
        Xf[base + e] = o;
        XH[base + e] = (_Float16)o;
    }
}

// ---------------------------------------------------------------------------
// Permute ul output rows: src[u*B+b] -> dst[b*U+u]  (the transpose(1,0,2))
// ---------------------------------------------------------------------------
__global__ __launch_bounds__(256)
void permute_kernel(const float* __restrict__ src, float* __restrict__ Xf,
                    _Float16* __restrict__ XH) {
    const int r = blockIdx.x;          // u*B + b
    const int u = r / kB, b = r % kB;
    const int dr = b * kU + u;
    for (int d = threadIdx.x; d < kD; d += 256) {
        const float v = src[(size_t)r * kD + d];
        Xf[(size_t)dr * kD + d] = v;
        XH[(size_t)dr * kD + d] = (_Float16)v;
    }
}

// ---------------------------------------------------------------------------
// Final linear [128, 24576] @ [24576, 97] + b  (0.6 GFLOP -> plain kernel)
// ---------------------------------------------------------------------------
__global__ __launch_bounds__(256)
void final_kernel(const float* __restrict__ X, const float* __restrict__ W,
                  const float* __restrict__ bias, float* __restrict__ out) {
    const int id = blockIdx.x * blockDim.x + threadIdx.x;
    if (id >= kB * kTGT) return;
    const int b = id / kTGT, t = id % kTGT;
    const float* x = X + (size_t)b * (kU * kD);
    float acc = bias[t];
    for (int k = 0; k < kU * kD; ++k) acc += x[k] * W[(size_t)k * kTGT + t];
    out[id] = acc;
}

// ---------------------------------------------------------------------------
// Host orchestration
// ---------------------------------------------------------------------------
extern "C" void kernel_launch(void* const* d_in, const int* in_sizes, int n_in,
                              void* d_out, int out_size, void* d_ws, size_t ws_size,
                              hipStream_t stream) {
    (void)in_sizes; (void)n_in; (void)out_size; (void)ws_size;

    const int*   i_sd   = (const int*)d_in[0];
    const int*   i_load = (const int*)d_in[1];
    const int*   i_st   = (const int*)d_in[2];
    const int*   i_sv   = (const int*)d_in[3];
    const int*   i_rt   = (const int*)d_in[4];
    const int*   i_pw   = (const int*)d_in[5];
    const int*   i_dw   = (const int*)d_in[6];
    const int*   i_vid  = (const int*)d_in[7];
    const int*   i_fl   = (const int*)d_in[8];
    const int*   i_dist = (const int*)d_in[9];
    const float* e_sd     = (const float*)d_in[10];
    const float* e_load   = (const float*)d_in[11];
    const float* e_status = (const float*)d_in[12];
    const float* e_served = (const float*)d_in[13];
    const float* e_rt     = (const float*)d_in[14];
    const float* e_time   = (const float*)d_in[15];
    const float* e_vid    = (const float*)d_in[16];
    const float* e_flag   = (const float*)d_in[17];
    const float* e_dist   = (const float*)d_in[18];
    const float* ue_Wq = (const float*)d_in[19]; const float* ue_bq = (const float*)d_in[20];
    const float* ue_Wk = (const float*)d_in[21]; const float* ue_bk = (const float*)d_in[22];
    const float* ue_Wv = (const float*)d_in[23]; const float* ue_bv = (const float*)d_in[24];
    const float* ue_Wo = (const float*)d_in[25]; const float* ue_bo = (const float*)d_in[26];
    const float* ue_W1 = (const float*)d_in[27]; const float* ue_b1 = (const float*)d_in[28];
    const float* ue_W2 = (const float*)d_in[29]; const float* ue_b2 = (const float*)d_in[30];
    const float* ue_g1 = (const float*)d_in[31]; const float* ue_be1 = (const float*)d_in[32];
    const float* ue_g2 = (const float*)d_in[33]; const float* ue_be2 = (const float*)d_in[34];
    const float* en_Wq = (const float*)d_in[35]; const float* en_bq = (const float*)d_in[36];
    const float* en_Wk = (const float*)d_in[37]; const float* en_bk = (const float*)d_in[38];
    const float* en_Wv = (const float*)d_in[39]; const float* en_bv = (const float*)d_in[40];
    const float* en_Wo = (const float*)d_in[41]; const float* en_bo = (const float*)d_in[42];
    const float* en_W1 = (const float*)d_in[43]; const float* en_b1 = (const float*)d_in[44];
    const float* en_W2 = (const float*)d_in[45]; const float* en_b2 = (const float*)d_in[46];
    const float* en_g1 = (const float*)d_in[47]; const float* en_be1 = (const float*)d_in[48];
    const float* en_g2 = (const float*)d_in[49]; const float* en_be2 = (const float*)d_in[50];
    const float* ul_W  = (const float*)d_in[51]; const float* ul_b  = (const float*)d_in[52];
    const float* fl_W  = (const float*)d_in[53]; const float* fl_b  = (const float*)d_in[54];

    // ---- workspace carve -------------------------------------------------
    char* ws = (char*)d_ws;
    size_t off = 0;
    auto take = [&](size_t bytes) -> void* {
        void* p = ws + off;
        off += (bytes + 255) & ~(size_t)255;
        return p;
    };
    float*    Xf  = (float*)   take((size_t)kM1 * kD * 4);
    _Float16* XH  = (_Float16*)take((size_t)kM1 * kD * 2);
    float*    TMP = (float*)   take((size_t)kM1 * kD * 4);
    _Float16* Qb  = (_Float16*)take((size_t)kM1 * kD * 2);
    _Float16* Kb  = (_Float16*)take((size_t)kM1 * kD * 2);
    _Float16* Vb  = (_Float16*)take((size_t)kM1 * kD * 2);
    _Float16* AO  = (_Float16*)take((size_t)kM1 * kD * 2);
    _Float16* HID = (_Float16*)take((size_t)kM1 * kDFF * 2);
    const size_t szQ = (size_t)2 * kD * kD, szF = (size_t)2 * kD * kDFF;
    _Float16* ueWq16 = (_Float16*)take(szQ * 2); _Float16* ueWk16 = (_Float16*)take(szQ * 2);
    _Float16* ueWv16 = (_Float16*)take(szQ * 2); _Float16* ueWo16 = (_Float16*)take(szQ * 2);
    _Float16* ueW116 = (_Float16*)take(szF * 2); _Float16* ueW216 = (_Float16*)take(szF * 2);
    const size_t szQ6 = (size_t)6 * kD * kD, szF6 = (size_t)6 * kD * kDFF;
    _Float16* enWq16 = (_Float16*)take(szQ6 * 2); _Float16* enWk16 = (_Float16*)take(szQ6 * 2);
    _Float16* enWv16 = (_Float16*)take(szQ6 * 2); _Float16* enWo16 = (_Float16*)take(szQ6 * 2);
    _Float16* enW116 = (_Float16*)take(szF6 * 2); _Float16* enW216 = (_Float16*)take(szF6 * 2);
    const size_t szUL = (size_t)kSU * kD * kD;
    _Float16* ulW16  = (_Float16*)take(szUL * 2);

    // ---- weight transpose-conversions (fp32 [K,N] -> fp16 [N,K]) ---------
    auto tcvt = [&](const float* s, _Float16* d, int Kd, int Nd, int layers) {
        const long total = (long)layers * Kd * Nd;
        tcvt_kernel<<<(unsigned)((total + 255) / 256), 256, 0, stream>>>(s, d, Kd, Nd, total);
    };
    tcvt(ue_Wq, ueWq16, kD, kD, 2);   tcvt(ue_Wk, ueWk16, kD, kD, 2);
    tcvt(ue_Wv, ueWv16, kD, kD, 2);   tcvt(ue_Wo, ueWo16, kD, kD, 2);
    tcvt(ue_W1, ueW116, kD, kDFF, 2); tcvt(ue_W2, ueW216, kDFF, kD, 2);
    tcvt(en_Wq, enWq16, kD, kD, 6);   tcvt(en_Wk, enWk16, kD, kD, 6);
    tcvt(en_Wv, enWv16, kD, kD, 6);   tcvt(en_Wo, enWo16, kD, kD, 6);
    tcvt(en_W1, enW116, kD, kDFF, 6); tcvt(en_W2, enW216, kDFF, kD, 6);
    tcvt(ul_W, ulW16, kSU * kD, kD, 1);

    // ---- GEMM launch helpers --------------------------------------------
    auto gemm_f16 = [&](const _Float16* A, const _Float16* W, const float* bias,
                        _Float16* C, int M, int N, int K, bool relu) {
        dim3 g(M / BM, N / BN);
        if (relu) gemm_wmma<true, false, true><<<g, 256, 0, stream>>>(A, W, bias, (float*)nullptr, C, M, N, K);
        else      gemm_wmma<false, false, true><<<g, 256, 0, stream>>>(A, W, bias, (float*)nullptr, C, M, N, K);
    };
    auto gemm_f32 = [&](const _Float16* A, const _Float16* W, const float* bias,
                        float* C, int M, int N, int K) {
        dim3 g(M / BM, N / BN);
        gemm_wmma<false, true, false><<<g, 256, 0, stream>>>(A, W, bias, C, (_Float16*)nullptr, M, N, K);
    };

    // ---- embedding gather ------------------------------------------------
    embed_kernel<<<kM1, 512, 0, stream>>>(i_sd, i_load, i_st, i_sv, i_rt, i_pw, i_dw,
                                          i_vid, i_fl, i_dist,
                                          e_sd, e_load, e_status, e_served, e_rt,
                                          e_time, e_vid, e_flag, e_dist, Xf, XH);

    // ---- user encoder: 2 layers, nseq=6144, S=19 ------------------------
    const size_t smem_u = (size_t)(3 * kSU * kDK + kSU * kSU) * 4;
    for (int L = 0; L < 2; ++L) {
        const size_t oQ = (size_t)L * kD * kD, oF1 = (size_t)L * kD * kDFF;
        const size_t oG = (size_t)L * kSU * kD;
        gemm_f16(XH, ueWq16 + oQ, ue_bq + L * kD, Qb, kM1, kD, kD, false);
        gemm_f16(XH, ueWk16 + oQ, ue_bk + L * kD, Kb, kM1, kD, kD, false);
        gemm_f16(XH, ueWv16 + oQ, ue_bv + L * kD, Vb, kM1, kD, kD, false);
        attn_kernel<<<kM2 * kH, 128, smem_u, stream>>>(Qb, Kb, Vb, AO, kSU);
        gemm_f32(AO, ueWo16 + oQ, ue_bo + L * kD, TMP, kM1, kD, kD);
        ln_kernel<<<kM2, 256, 0, stream>>>(Xf, XH, TMP, ue_g1 + oG, ue_be1 + oG, kSU);
        gemm_f16(XH, ueW116 + oF1, ue_b1 + L * kDFF, HID, kM1, kDFF, kD, true);
        gemm_f32(HID, ueW216 + oF1, ue_b2 + L * kD, TMP, kM1, kD, kDFF);
        ln_kernel<<<kM2, 256, 0, stream>>>(Xf, XH, TMP, ue_g2 + oG, ue_be2 + oG, kSU);
    }

    // ---- ul linear: [6144, 19*512] @ [9728, 512], then u<->b permute -----
    gemm_f32(XH, ulW16, ul_b, TMP, kM2, kD, kSU * kD);
    permute_kernel<<<kM2, 256, 0, stream>>>(TMP, Xf, XH);

    // ---- main encoder: 6 layers, nseq=128, S=48 --------------------------
    const size_t smem_m = (size_t)(3 * kU * kDK + kU * kU) * 4;
    for (int L = 0; L < 6; ++L) {
        const size_t oQ = (size_t)L * kD * kD, oF1 = (size_t)L * kD * kDFF;
        const size_t oG = (size_t)L * kU * kD;
        gemm_f16(XH, enWq16 + oQ, en_bq + L * kD, Qb, kM2, kD, kD, false);
        gemm_f16(XH, enWk16 + oQ, en_bk + L * kD, Kb, kM2, kD, kD, false);
        gemm_f16(XH, enWv16 + oQ, en_bv + L * kD, Vb, kM2, kD, kD, false);
        attn_kernel<<<kB * kH, 128, smem_m, stream>>>(Qb, Kb, Vb, AO, kU);
        gemm_f32(AO, enWo16 + oQ, en_bo + L * kD, TMP, kM2, kD, kD);
        ln_kernel<<<kB, 256, 0, stream>>>(Xf, XH, TMP, en_g1 + oG, en_be1 + oG, kU);
        gemm_f16(XH, enW116 + oF1, en_b1 + L * kDFF, HID, kM2, kDFF, kD, true);
        gemm_f32(HID, enW216 + oF1, en_b2 + L * kD, TMP, kM2, kD, kDFF);
        ln_kernel<<<kB, 256, 0, stream>>>(Xf, XH, TMP, en_g2 + oG, en_be2 + oG, kU);
    }

    // ---- final linear -> d_out [128, 97] ---------------------------------
    final_kernel<<<(kB * kTGT + 255) / 256, 256, 0, stream>>>(Xf, fl_W, fl_b, (float*)d_out);
}
